// Network_16234976379418
// MI455X (gfx1250) — matrix operand order repaired and checked by reading, not compiled, not run on hardware
//
#include <hip/hip_runtime.h>
#include <hip/hip_bf16.h>

constexpr int VOCAB  = 50000;
constexpr int E      = 512;
constexpr int H      = 1024;
constexpr int BB     = 128;   // batch
constexpr int SS     = 256;   // seq len
constexpr int MLP_H  = 64;
constexpr int NCLASS = 5;

typedef __attribute__((ext_vector_type(16))) __bf16 v16bf;
typedef __attribute__((ext_vector_type(8)))  __bf16 v8bf;
typedef __attribute__((ext_vector_type(8)))  float  v8f;

#if __has_builtin(__builtin_amdgcn_sched_barrier)
#define SCHED_FENCE() __builtin_amdgcn_sched_barrier(0)
#else
#define SCHED_FENCE()
#endif

// ---- fragment K-index maps used here (wave32; A/B k order is this code's contiguous choice) ----
// A (16x32 bf16): element e of lane -> K = (lane/16)*16 + e,              row M = lane&15
//   -> bf16 row-major A: two contiguous 16B loads per k-step
// B (32x16 bf16): element e of lane -> K = (lane/16)*16 + e,                col N = lane&15
//   -> bf16 col-major B (Bt[n][k]): one contiguous 32B load per k-step
// C/D (16x16 f32): vgpr v of lane   -> M = v + 8*(lane/16),                 col N = lane&15

__device__ inline v16bf load_afrag(const __bf16* ap) {
  v8bf alo = *(const v8bf*)(ap);
  v8bf ahi = *(const v8bf*)(ap + 8);
  return __builtin_shufflevector(alo, ahi, 0, 1, 2, 3, 4, 5, 6, 7,
                                 8, 9, 10, 11, 12, 13, 14, 15);
}

__device__ inline v8f wmma_bf(v16bf a, v16bf b, v8f c) {
  return __builtin_amdgcn_wmma_f32_16x16x32_bf16(false, a, false, b,
                                                 (short)0, c, false, false);
}

// 16x16 tile: A row-major bf16, B col-major bf16 (bcol = Bt + n*K).
__device__ inline v8f tile_gemm_bfbf(const __bf16* __restrict__ arow,
                                     const __bf16* __restrict__ bcol,
                                     int K, int half, v8f acc) {
  for (int k0 = 0; k0 < K; k0 += 32) {
    v16bf a = load_afrag(arow + k0 + (half << 4));
    v16bf b = *(const v16bf*)(bcol + k0 + (half << 4));
    acc = wmma_bf(a, b, acc);
  }
  return acc;
}

// 16x64 tile, double-buffered with a scheduling fence so the backend cannot
// sink next-iteration loads below the current wmmas: emitted order is
// [10 b128 loads for k+32] | fence | [4 wmmas on k] -> wmmas wait only on
// loads issued a full iteration earlier (true per-wave pipeline).
__device__ inline void tile_gemm_bfbf_x4(const __bf16* __restrict__ arow,
                                         const __bf16* __restrict__ bc0,
                                         const __bf16* __restrict__ bc1,
                                         const __bf16* __restrict__ bc2,
                                         const __bf16* __restrict__ bc3,
                                         int K, int half, v8f acc[4]) {
  const int ao = half << 4;
  const int bo = half << 4;
  v16bf a  = load_afrag(arow + ao);
  v16bf b0 = *(const v16bf*)(bc0 + bo);
  v16bf b1 = *(const v16bf*)(bc1 + bo);
  v16bf b2 = *(const v16bf*)(bc2 + bo);
  v16bf b3 = *(const v16bf*)(bc3 + bo);
  for (int k0 = 32; k0 < K; k0 += 32) {
    v16bf an  = load_afrag(arow + k0 + ao);
    v16bf bn0 = *(const v16bf*)(bc0 + k0 + bo);
    v16bf bn1 = *(const v16bf*)(bc1 + k0 + bo);
    v16bf bn2 = *(const v16bf*)(bc2 + k0 + bo);
    v16bf bn3 = *(const v16bf*)(bc3 + k0 + bo);
    SCHED_FENCE();                 // keep next-iter loads above current wmmas
    acc[0] = wmma_bf(a, b0, acc[0]);
    acc[1] = wmma_bf(a, b1, acc[1]);
    acc[2] = wmma_bf(a, b2, acc[2]);
    acc[3] = wmma_bf(a, b3, acc[3]);
    SCHED_FENCE();                 // keep wmmas above the following iteration
    a = an; b0 = bn0; b1 = bn1; b2 = bn2; b3 = bn3;
  }
  acc[0] = wmma_bf(a, b0, acc[0]);
  acc[1] = wmma_bf(a, b1, acc[1]);
  acc[2] = wmma_bf(a, b2, acc[2]);
  acc[3] = wmma_bf(a, b3, acc[3]);
}

// f32 A with on-the-fly cvt (tiny MLP1 only); bf16 col-major B.
__device__ inline v8f tile_gemm_f32a(const float* __restrict__ arow,
                                     const __bf16* __restrict__ bcol,
                                     int K, int half, v8f acc) {
  for (int k0 = 0; k0 < K; k0 += 32) {
    v16bf a;
#pragma unroll
    for (int e = 0; e < 16; ++e) {
      const int ka = (half << 4) + e;
      a[e] = (__bf16)arow[k0 + ka];
    }
    v16bf b = *(const v16bf*)(bcol + k0 + (half << 4));
    acc = wmma_bf(a, b, acc);
  }
  return acc;
}

// ---------------- operand pre-conversion ----------------
// out[n*K + k] = (bf16) in[k*N + n]   (transpose + convert; contiguous writes)
__global__ void k_cvt_transpose(const float* __restrict__ in, __bf16* __restrict__ out,
                                int K, int N) {
  size_t i = (size_t)blockIdx.x * blockDim.x + threadIdx.x;
  if (i < (size_t)K * N) {
    int n = (int)(i / K), k = (int)(i % K);
    out[i] = (__bf16)in[(size_t)k * N + n];
  }
}

// straight f32 -> bf16 copy (embedding table)
__global__ void k_cvt(const float* __restrict__ in, __bf16* __restrict__ out, size_t count) {
  size_t i = (size_t)blockIdx.x * blockDim.x + threadIdx.x;
  if (i < count) out[i] = (__bf16)in[i];
}

// ---------------- init state ----------------
__global__ void k_init(float* h, __bf16* hb, float* hsum) {
  int i = blockIdx.x * blockDim.x + threadIdx.x;
  if (i < BB * H) { h[i] = 0.f; hb[i] = (__bf16)0.f; hsum[i] = 0.f; }
}

// ---------------- fused embedding-gather + X @ W[:E] GEMM (16x64/wave) ----------
// out[(s*BB+b), n] = emb[sent[b,s], :] @ W[0:E, n]   (bias added later)
__global__ __launch_bounds__(256) void k_embed_gemm(const int* __restrict__ sent,
                                                    const __bf16* __restrict__ embb,
                                                    const __bf16* __restrict__ Wt,  // [H][E]
                                                    float* __restrict__ out) {
  const int lane = threadIdx.x & 31, wv = threadIdx.x >> 5;
  const int mt = blockIdx.y * 8 + wv;          // 0..2047  (M tiles of S*B)
  const int ng = blockIdx.x;                   // 0..15    (64-wide N groups of H)
  const int half = lane >> 4, mr = lane & 15;
  const int t = mt * 16 + mr;                  // row = s*BB + b
  const int s = t >> 7, b = t & (BB - 1);
  const int vrow = sent[b * SS + s];
  const int n0 = ng * 64 + mr;                 // columns n0, n0+16, n0+32, n0+48
  v8f acc[4] = {{}, {}, {}, {}};
  tile_gemm_bfbf_x4(embb + (size_t)vrow * E,
                    Wt + (size_t)(n0)      * E,
                    Wt + (size_t)(n0 + 16) * E,
                    Wt + (size_t)(n0 + 32) * E,
                    Wt + (size_t)(n0 + 48) * E,
                    E, half, acc);
#pragma unroll
  for (int j = 0; j < 4; ++j) {
#pragma unroll
    for (int v = 0; v < 8; ++v) {
      const int row = mt * 16 + half * 8 + v;
      out[(size_t)row * H + (n0 + 16 * j)] = acc[j][v];
    }
  }
}

// ---------------- per-step: z and r gates, rh = r*h ----------------
__global__ __launch_bounds__(256) void k_gates(const __bf16* __restrict__ hb,
                                               const float* __restrict__ h,
                                               const __bf16* __restrict__ Uzt, // [H][H]
                                               const __bf16* __restrict__ Urt, // [H][H]
                                               const float* __restrict__ xz,
                                               const float* __restrict__ xr,
                                               const float* __restrict__ bz,
                                               const float* __restrict__ br,
                                               float* __restrict__ zout,
                                               __bf16* __restrict__ rhout, int step) {
  const int lane = threadIdx.x & 31, wv = threadIdx.x >> 5;
  const int wid  = blockIdx.x * 8 + wv;        // 0..1023
  const int gate = wid >> 9;                   // 0: z, 1: r
  const int tile = wid & 511;
  const int mt = tile >> 6, nt = tile & 63;
  const int half = lane >> 4, mr = lane & 15;
  const int m = mt * 16 + mr, n = nt * 16 + mr;
  const __bf16* Ut  = (gate == 0 ? Uzt : Urt);
  const float* xg   = (gate == 0 ? xz : xr) + (size_t)step * BB * H;
  const float* bias = (gate == 0 ? bz : br);
  v8f acc = {};
  acc = tile_gemm_bfbf(hb + (size_t)m * H, Ut + (size_t)n * H, H, half, acc);
#pragma unroll
  for (int v = 0; v < 8; ++v) {
    const int row = mt * 16 + half * 8 + v;
    const float pre = acc[v] + xg[(size_t)row * H + n] + bias[n];
    const float sg  = 1.f / (1.f + __expf(-pre));
    if (gate == 0) zout[(size_t)row * H + n] = sg;
    else           rhout[(size_t)row * H + n] = (__bf16)(sg * h[(size_t)row * H + n]);
  }
}

// ---------------- per-step: h_hat, h update, running sum ----------------
__global__ __launch_bounds__(256) void k_update(const __bf16* __restrict__ rhb,
                                                const __bf16* __restrict__ Umt, // [H][H]
                                                const float* __restrict__ xm,
                                                const float* __restrict__ bm,
                                                const float* __restrict__ zbuf,
                                                float* __restrict__ h,
                                                __bf16* __restrict__ hb,
                                                float* __restrict__ hsum, int step) {
  const int lane = threadIdx.x & 31, wv = threadIdx.x >> 5;
  const int tile = blockIdx.x * 8 + wv;        // 0..511
  const int mt = tile >> 6, nt = tile & 63;
  const int half = lane >> 4, mr = lane & 15;
  const int m = mt * 16 + mr, n = nt * 16 + mr;
  const float* xg = xm + (size_t)step * BB * H;
  v8f acc = {};
  acc = tile_gemm_bfbf(rhb + (size_t)m * H, Umt + (size_t)n * H, H, half, acc);
#pragma unroll
  for (int v = 0; v < 8; ++v) {
    const int row = mt * 16 + half * 8 + v;
    const size_t idx = (size_t)row * H + n;
    const float pre  = acc[v] + xg[idx] + bm[n];
    const float hh   = tanhf(pre);
    const float zv   = zbuf[idx];
    const float hold = h[idx];
    const float hn   = hold + zv * (hh - hold);   // (1-z)*h + z*h_hat
    h[idx]    = hn;
    hb[idx]   = (__bf16)hn;
    hsum[idx] += hn;
  }
}

// ---------------- MLP layer 1: relu((hsum/S) @ W1 + b1) ----------------
__global__ __launch_bounds__(256) void k_mlp1(const float* __restrict__ hsum,
                                              const __bf16* __restrict__ W1t, // [64][H]
                                              const float* __restrict__ b1,
                                              float* __restrict__ act) {
  const int lane = threadIdx.x & 31, wv = threadIdx.x >> 5;   // wv = M tile (8)
  const int half = lane >> 4, mr = lane & 15;
  const float invS = 1.f / (float)SS;
  const int m = wv * 16 + mr;
  for (int nt = 0; nt < MLP_H / 16; ++nt) {
    const int n = nt * 16 + mr;
    v8f acc = {};
    acc = tile_gemm_f32a(hsum + (size_t)m * H, W1t + (size_t)n * H, H, half, acc);
#pragma unroll
    for (int v = 0; v < 8; ++v) {
      const int row = wv * 16 + half * 8 + v;
      const float r = acc[v] * invS + b1[n];     // mean applied post-GEMM
      act[row * MLP_H + n] = r > 0.f ? r : 0.f;
    }
  }
}

// ---------------- MLP layer 2 (tiny): act @ W2 + b2 ----------------
__global__ void k_mlp2(const float* __restrict__ act, const float* __restrict__ W2,
                       const float* __restrict__ b2, float* __restrict__ out) {
  const int i = blockIdx.x * blockDim.x + threadIdx.x;
  if (i < BB * NCLASS) {
    const int b = i / NCLASS, c = i % NCLASS;
    float s = b2[c];
    for (int k = 0; k < MLP_H; ++k) s += act[b * MLP_H + k] * W2[k * NCLASS + c];
    out[i] = s;
  }
}

extern "C" void kernel_launch(void* const* d_in, const int* in_sizes, int n_in,
                              void* d_out, int out_size, void* d_ws, size_t ws_size,
                              hipStream_t stream) {
  const int*   sent = (const int*)d_in[0];
  const float* emb  = (const float*)d_in[2];
  const float* Wz   = (const float*)d_in[3];
  const float* bz   = (const float*)d_in[4];
  const float* Wr   = (const float*)d_in[5];
  const float* br   = (const float*)d_in[6];
  const float* Wm   = (const float*)d_in[7];
  const float* bm   = (const float*)d_in[8];
  const float* W1   = (const float*)d_in[9];
  const float* b1   = (const float*)d_in[10];
  const float* W2   = (const float*)d_in[11];
  const float* b2   = (const float*)d_in[12];
  float* out = (float*)d_out;

  char* p = (char*)d_ws;
  const size_t SBH = (size_t)SS * BB * H;
  const size_t BH  = (size_t)BB * H;
  float*  xz   = (float*)p;  p += SBH * sizeof(float);
  float*  xr   = (float*)p;  p += SBH * sizeof(float);
  float*  xm   = (float*)p;  p += SBH * sizeof(float);
  float*  h    = (float*)p;  p += BH * sizeof(float);
  float*  hsum = (float*)p;  p += BH * sizeof(float);
  float*  zb   = (float*)p;  p += BH * sizeof(float);
  float*  act  = (float*)p;  p += (size_t)BB * MLP_H * sizeof(float);
  __bf16* hb   = (__bf16*)p; p += BH * sizeof(__bf16);
  __bf16* rhb  = (__bf16*)p; p += BH * sizeof(__bf16);
  __bf16* embb = (__bf16*)p; p += (size_t)VOCAB * E * sizeof(__bf16);
  __bf16* Wzxt = (__bf16*)p; p += (size_t)H * E * sizeof(__bf16);  // [H][E]
  __bf16* Wrxt = (__bf16*)p; p += (size_t)H * E * sizeof(__bf16);
  __bf16* Wmxt = (__bf16*)p; p += (size_t)H * E * sizeof(__bf16);
  __bf16* Uzt  = (__bf16*)p; p += (size_t)H * H * sizeof(__bf16);  // [H][H]
  __bf16* Urt  = (__bf16*)p; p += (size_t)H * H * sizeof(__bf16);
  __bf16* Umt  = (__bf16*)p; p += (size_t)H * H * sizeof(__bf16);
  __bf16* W1t  = (__bf16*)p; p += (size_t)MLP_H * H * sizeof(__bf16);

  // ---- one-time per-call operand conversion (streaming, ~120 MB total) ----
  {
    const size_t ne = (size_t)VOCAB * E;
    k_cvt<<<(unsigned)((ne + 255) / 256), 256, 0, stream>>>(emb, embb, ne);
    const unsigned gEH = (unsigned)(((size_t)E * H + 255) / 256);
    const unsigned gHH = (unsigned)(((size_t)H * H + 255) / 256);
    k_cvt_transpose<<<gEH, 256, 0, stream>>>(Wz, Wzxt, E, H);            // rows 0..E-1
    k_cvt_transpose<<<gEH, 256, 0, stream>>>(Wr, Wrxt, E, H);
    k_cvt_transpose<<<gEH, 256, 0, stream>>>(Wm, Wmxt, E, H);
    k_cvt_transpose<<<gHH, 256, 0, stream>>>(Wz + (size_t)E * H, Uzt, H, H);
    k_cvt_transpose<<<gHH, 256, 0, stream>>>(Wr + (size_t)E * H, Urt, H, H);
    k_cvt_transpose<<<gHH, 256, 0, stream>>>(Wm + (size_t)E * H, Umt, H, H);
    const unsigned g1 = (unsigned)(((size_t)H * MLP_H + 255) / 256);
    k_cvt_transpose<<<g1, 256, 0, stream>>>(W1, W1t, H, MLP_H);
  }

  // ---- hoisted input-side GEMMs: x_seq @ W[:E]  (one per gate) ----
  dim3 egrid(H / 64, (SS * BB / 16) / 8);   // (16, 256), 16x64 tile per wave
  k_embed_gemm<<<egrid, 256, 0, stream>>>(sent, embb, Wzxt, xz);
  k_embed_gemm<<<egrid, 256, 0, stream>>>(sent, embb, Wrxt, xr);
  k_embed_gemm<<<egrid, 256, 0, stream>>>(sent, embb, Wmxt, xm);

  k_init<<<(BB * H + 255) / 256, 256, 0, stream>>>(h, hb, hsum);

  // ---- sequential recurrence: bf16 recurrent weights L2-resident (6 MB) ----
  for (int st = 0; st < SS; ++st) {
    k_gates<<<128, 256, 0, stream>>>(hb, h, Uzt, Urt, xz, xr, bz, br, zb, rhb, st);
    k_update<<<64, 256, 0, stream>>>(rhb, Umt, xm, bm, zb, h, hb, hsum, st);
  }

  k_mlp1<<<1, 256, 0, stream>>>(hsum, W1t, b1, act);
  k_mlp2<<<5, 128, 0, stream>>>(act, W2, b2, out);
}
